// SparseMultiHeadAttention_62294205661288
// MI455X (gfx1250) — matrix operand order repaired and checked by reading, not compiled, not run on hardware
//
#include <hip/hip_runtime.h>

// ---- problem constants (from reference) ----
#define B_   4
#define S_   2048
#define D_   1024
#define H_   16
#define HD_  64
#define M_   (B_ * S_)        // 8192 rows for all projections

// ---- GEMM tiling ----
#define BM 128
#define BN 128
#define BK 32
#define KTOT 1024
#define NTOT 1024

typedef __attribute__((ext_vector_type(16))) __bf16 v16bf;
typedef __attribute__((ext_vector_type(16))) short  v16s;
typedef __attribute__((ext_vector_type(8)))  short  v8s;
typedef __attribute__((ext_vector_type(8)))  float  v8f;

__device__ __forceinline__ unsigned short f2bf(float f) {
  unsigned int u = __float_as_uint(f);
  u += 0x7FFFu + ((u >> 16) & 1u);          // round-to-nearest-even
  return (unsigned short)(u >> 16);
}
__device__ __forceinline__ float bf2f(unsigned short h) {
  return __uint_as_float(((unsigned int)h) << 16);
}

// ---- CDNA5 async global->LDS copy (ASYNCcnt-tracked), 16B per lane ----
__device__ __forceinline__ void async_b128(void* lds, const void* g) {
  unsigned l = (unsigned)(uintptr_t)lds;                  // low 32 bits = LDS byte address
  unsigned long long ga = (unsigned long long)(uintptr_t)g;
  asm volatile("global_load_async_to_lds_b128 %0, %1, off"
               :: "v"(l), "v"(ga) : "memory");
}
__device__ __forceinline__ void wait_async0() {
#if __has_builtin(__builtin_amdgcn_s_wait_asynccnt)
  __builtin_amdgcn_s_wait_asynccnt(0);
#else
  asm volatile("s_wait_asynccnt 0x0" ::: "memory");
#endif
}

// ---------------------------------------------------------------------------
// f32 -> bf16 conversion, 4 elements / thread (for x)
// ---------------------------------------------------------------------------
__global__ __launch_bounds__(256)
void cvt4(const float* __restrict__ in, unsigned short* __restrict__ out, int n4) {
  int i = blockIdx.x * blockDim.x + threadIdx.x;
  if (i < n4) {
    float4 v = reinterpret_cast<const float4*>(in)[i];
    unsigned int lo = (unsigned int)f2bf(v.x) | ((unsigned int)f2bf(v.y) << 16);
    unsigned int hi = (unsigned int)f2bf(v.z) | ((unsigned int)f2bf(v.w) << 16);
    reinterpret_cast<uint2*>(out)[i] = make_uint2(lo, hi);
  }
}

// ---------------------------------------------------------------------------
// f32 [K][N] -> bf16 transposed [N][K] (for the weight matrices)
// 32x32 tiles through LDS; coalesced on both sides.
// ---------------------------------------------------------------------------
__global__ __launch_bounds__(256)
void cvtT(const float* __restrict__ in, unsigned short* __restrict__ out) {
  __shared__ float tile[32][33];
  const int t  = threadIdx.x;
  const int c  = t & 31;        // inner (coalesced) index
  const int rg = t >> 5;        // 0..7 -> 4 rows each
  const int k0 = blockIdx.y * 32;
  const int n0 = blockIdx.x * 32;
  #pragma unroll
  for (int i = 0; i < 4; ++i) {
    int k = rg * 4 + i;
    tile[k][c] = in[(size_t)(k0 + k) * NTOT + n0 + c];
  }
  __syncthreads();
  #pragma unroll
  for (int i = 0; i < 4; ++i) {
    int n = rg * 4 + i;
    out[(size_t)(n0 + n) * KTOT + k0 + c] = f2bf(tile[c][n]);
  }
}

// ---------------------------------------------------------------------------
// C[M,N] = A[M,K](bf16) @ Bt[N,K](bf16)^T + bias[N], f32 accum via WMMA.
// Double-buffered LDS tiles filled with async global->LDS copies.
// 256 threads = 8 waves; wave -> 64x32 slab = 4x2 WMMA tiles.
// ---------------------------------------------------------------------------
__global__ __launch_bounds__(256)
void gemm_bf16_wmma(const unsigned short* __restrict__ A,
                    const unsigned short* __restrict__ Bt,
                    const float* __restrict__ bias,
                    float* __restrict__ Cf,
                    unsigned short* __restrict__ Cb)
{
  __shared__ unsigned short As[2][BM][BK];   // [m][k], 8 KB per buffer
  __shared__ unsigned short Bs[2][BN][BK];   // [n][k], 8 KB per buffer

  const int tid  = threadIdx.x;
  const int lane = tid & 31;
  const int wid  = tid >> 5;     // 0..7
  const int wm   = wid & 1;      // row slab: 0..1  (64 rows each)
  const int wn   = wid >> 1;     // col slab: 0..3  (32 cols each)
  const int l16  = lane & 15;
  const int lh   = lane >> 4;    // half-wave select

  const int gm0 = blockIdx.y * BM;
  const int gn0 = blockIdx.x * BN;

  v8f acc[4][2];
  #pragma unroll
  for (int im = 0; im < 4; ++im)
    #pragma unroll
    for (int in = 0; in < 2; ++in)
      #pragma unroll
      for (int r = 0; r < 8; ++r) acc[im][in][r] = 0.0f;

  // stage one 128x32 bf16 tile pair (A and Bt are the same shape/layout)
  auto stage = [&](int buf, int kt) {
    #pragma unroll
    for (int u = tid; u < (BM * BK) / 8; u += 256) {   // 512 16B units each
      int r = u >> 2;
      int c = (u & 3) * 8;
      async_b128(&As[buf][r][c], A  + (size_t)(gm0 + r) * KTOT + kt + c);
      async_b128(&Bs[buf][r][c], Bt + (size_t)(gn0 + r) * KTOT + kt + c);
    }
  };

  stage(0, 0);
  wait_async0();
  __syncthreads();

  int cur = 0;
  #pragma unroll 1
  for (int kt = 0; kt < KTOT; kt += BK) {
    const int nxt = cur ^ 1;
    if (kt + BK < KTOT) stage(nxt, kt + BK);   // overlap next tile with compute

    // ---- fragments per documented CDNA5 VGPR layouts ----
    v16bf afrag[4];
    #pragma unroll
    for (int im = 0; im < 4; ++im) {
      int mr = wm * 64 + im * 16 + l16;
      v8s lo = *reinterpret_cast<const v8s*>(&As[cur][mr][lh * 8]);       // K = h*8 .. +7
      v8s hi = *reinterpret_cast<const v8s*>(&As[cur][mr][16 + lh * 8]);  // K = 16+h*8 .. +7
      v16s t = __builtin_shufflevector(lo, hi, 0,1,2,3,4,5,6,7,8,9,10,11,12,13,14,15);
      afrag[im] = __builtin_bit_cast(v16bf, t);
    }
    v16bf bfrag[2];
    #pragma unroll
    for (int in = 0; in < 2; ++in) {
      int nr = wn * 32 + in * 16 + l16;
      v8s lo = *reinterpret_cast<const v8s*>(&Bs[cur][nr][lh * 8]);       // K = h*8 .. +7
      v8s hi = *reinterpret_cast<const v8s*>(&Bs[cur][nr][16 + lh * 8]);  // K = 16+h*8 .. +7
      v16s t = __builtin_shufflevector(lo, hi, 0,1,2,3,4,5,6,7,8,9,10,11,12,13,14,15);
      bfrag[in] = __builtin_bit_cast(v16bf, t);
    }

    #pragma unroll
    for (int im = 0; im < 4; ++im)
      #pragma unroll
      for (int in = 0; in < 2; ++in)
        acc[im][in] = __builtin_amdgcn_wmma_f32_16x16x32_bf16(
            false, afrag[im], false, bfrag[in],
            (short)0, acc[im][in], false, false);

    wait_async0();       // own async copies into nxt complete
    __syncthreads();     // everyone done reading cur / writing nxt
    cur = nxt;
  }

  // ---- epilogue: bias + store (f32 or bf16) ----
  #pragma unroll
  for (int in = 0; in < 2; ++in) {
    int n = gn0 + wn * 32 + in * 16 + l16;
    float bv = bias[n];
    #pragma unroll
    for (int im = 0; im < 4; ++im) {
      #pragma unroll
      for (int r = 0; r < 8; ++r) {
        int m = gm0 + wm * 64 + im * 16 + lh * 8 + r;   // C/D layout: VGPR r -> M=r (+8 hi half)
        float val = acc[im][in][r] + bv;
        if (Cf) Cf[(size_t)m * NTOT + n] = val;
        else    Cb[(size_t)m * NTOT + n] = f2bf(val);
      }
    }
  }
}

// ---------------------------------------------------------------------------
// Windowed attention: one wave (32 lanes) per (b,h,q) row.
// Writes the full dense weights row (zeros outside band) + bf16 context.
// ---------------------------------------------------------------------------
__global__ __launch_bounds__(256)
void attn_window(const unsigned short* __restrict__ qb,
                 const unsigned short* __restrict__ kb,
                 const unsigned short* __restrict__ vb,
                 float* __restrict__ wout,
                 unsigned short* __restrict__ ctx)
{
  const int lane = threadIdx.x & 31;
  const int row  = (int)((blockIdx.x * blockDim.x + threadIdx.x) >> 5); // global wave id
  const int q = row & (S_ - 1);
  const int h = (row >> 11) & (H_ - 1);
  const int b = row >> 15;

  const size_t base = ((size_t)b * S_) * D_ + (size_t)h * HD_;
  const unsigned short* qrow = qb + base + (size_t)q * D_;

  // lanes 0..8 own window offsets -4..+4
  float sc = -__builtin_inff();
  int j = q - 4 + lane;
  if (lane < 9 && j >= 0 && j < S_) {
    const unsigned short* krow = kb + base + (size_t)j * D_;
    float acc = 0.f;
    #pragma unroll
    for (int d = 0; d < HD_; d += 8) {
      uint4 qv = *reinterpret_cast<const uint4*>(qrow + d);
      uint4 kv = *reinterpret_cast<const uint4*>(krow + d);
      const unsigned short* qp = (const unsigned short*)&qv;
      const unsigned short* kp = (const unsigned short*)&kv;
      #pragma unroll
      for (int e = 0; e < 8; ++e) acc += bf2f(qp[e]) * bf2f(kp[e]);
    }
    sc = acc * 0.125f;                     // 1/sqrt(64)
  }

  // softmax over the wave (inactive lanes contribute -inf / 0)
  float mx = sc;
  #pragma unroll
  for (int off = 16; off; off >>= 1) mx = fmaxf(mx, __shfl_xor(mx, off));
  float ev = (sc > -__builtin_inff()) ? __expf(sc - mx) : 0.f;
  float sum = ev;
  #pragma unroll
  for (int off = 16; off; off >>= 1) sum += __shfl_xor(sum, off);
  float w = ev / sum;

  // ---- stream full weights row: 16 chunks x 128 floats, b128 stores ----
  float* wrow = wout + (size_t)row * S_;
  const int lo = q - 4, hi = q + 4;
  #pragma unroll 1
  for (int c = 0; c < S_; c += 128) {
    float4 val = make_float4(0.f, 0.f, 0.f, 0.f);
    if (c <= hi && c + 127 >= lo) {        // wave-uniform band check
      int j0 = c + lane * 4;
      float* vp = (float*)&val;
      #pragma unroll
      for (int t = 0; t < 4; ++t) {
        int idx = j0 + t - lo;             // lane index holding this weight
        float ws = __shfl(w, idx & 31);
        vp[t] = ((unsigned)idx < 9u) ? ws : 0.f;
      }
    }
    *reinterpret_cast<float4*>(wrow + c + lane * 4) = val;
  }

  // ---- context: 64 dims, 2 per lane, packed bf16 stores ----
  float a0 = 0.f, a1 = 0.f;
  const int d0 = lane * 2;
  const unsigned short* vbase = vb + base;
  #pragma unroll
  for (int i = 0; i < 9; ++i) {
    int jj = q - 4 + i;
    float wi = __shfl(w, i);
    if (jj >= 0 && jj < S_) {
      unsigned int pk = *reinterpret_cast<const unsigned int*>(vbase + (size_t)jj * D_ + d0);
      a0 += wi * bf2f((unsigned short)(pk & 0xFFFFu));
      a1 += wi * bf2f((unsigned short)(pk >> 16));
    }
  }
  unsigned int opk = (unsigned int)f2bf(a0) | ((unsigned int)f2bf(a1) << 16);
  *reinterpret_cast<unsigned int*>(ctx + base + (size_t)q * D_ + d0) = opk;
}

// ---------------------------------------------------------------------------
extern "C" void kernel_launch(void* const* d_in, const int* in_sizes, int n_in,
                              void* d_out, int out_size, void* d_ws, size_t ws_size,
                              hipStream_t stream) {
  const float* x  = (const float*)d_in[0];
  const float* Wq = (const float*)d_in[1];
  const float* bq = (const float*)d_in[2];
  const float* Wk = (const float*)d_in[3];
  const float* bk = (const float*)d_in[4];
  const float* Wv = (const float*)d_in[5];
  const float* bv = (const float*)d_in[6];
  const float* Wo = (const float*)d_in[7];
  const float* bo = (const float*)d_in[8];

  const size_t MB = 1u << 20;
  char* ws = (char*)d_ws;
  unsigned short* xb   = (unsigned short*)(ws);              // 16 MB, reused as ctx
  unsigned short* Wqb  = (unsigned short*)(ws + 16 * MB);    //  2 MB each, TRANSPOSED [N][K]
  unsigned short* Wkb  = (unsigned short*)(ws + 18 * MB);
  unsigned short* Wvb  = (unsigned short*)(ws + 20 * MB);
  unsigned short* Wob  = (unsigned short*)(ws + 22 * MB);
  unsigned short* qbuf = (unsigned short*)(ws + 24 * MB);    // 16 MB each
  unsigned short* kbuf = (unsigned short*)(ws + 40 * MB);
  unsigned short* vbuf = (unsigned short*)(ws + 56 * MB);    // total 72 MB

  float* out = (float*)d_out;                       // [B,S,D] f32
  float* wgt = out + (size_t)M_ * D_;               // [B,H,S,S] f32

  dim3 blk(256);

  // x: straight f32->bf16; W*: f32->bf16 + transpose to [N][K]
  cvt4<<<(M_ * D_ / 4 + 255) / 256, blk, 0, stream>>>(x, xb, M_ * D_ / 4);
  dim3 gt(NTOT / 32, KTOT / 32);
  cvtT<<<gt, blk, 0, stream>>>(Wq, Wqb);
  cvtT<<<gt, blk, 0, stream>>>(Wk, Wkb);
  cvtT<<<gt, blk, 0, stream>>>(Wv, Wvb);
  cvtT<<<gt, blk, 0, stream>>>(Wo, Wob);

  // Q/K/V projections (bf16 out)
  dim3 g(NTOT / BN, M_ / BM);   // (8, 64)
  gemm_bf16_wmma<<<g, blk, 0, stream>>>(xb, Wqb, bq, nullptr, qbuf);
  gemm_bf16_wmma<<<g, blk, 0, stream>>>(xb, Wkb, bk, nullptr, kbuf);
  gemm_bf16_wmma<<<g, blk, 0, stream>>>(xb, Wvb, bv, nullptr, vbuf);

  // banded attention: weights (f32, dense) + context (bf16, reuses xb)
  attn_window<<<(B_ * H_ * S_) / 8, blk, 0, stream>>>(qbuf, kbuf, vbuf, wgt, xb);

  // output projection (f32 out)
  gemm_bf16_wmma<<<g, blk, 0, stream>>>(xb, Wob, bo, out, nullptr);
}
